// SequenceCompressorResampler_21706764714100
// MI455X (gfx1250) — compile-verified, matrix operand order checked by reading, not run on hardware
//
#include <hip/hip_runtime.h>
#include <hip/hip_bf16.h>
#include <math.h>

// ---------------------------------------------------------------------------
// Types / helpers
// ---------------------------------------------------------------------------
typedef unsigned short u16t;
typedef unsigned int   u32t;
typedef __bf16 bf16t;
typedef bf16t v16bf __attribute__((ext_vector_type(16)));
typedef float v8f  __attribute__((ext_vector_type(8)));

#if defined(__has_builtin)
#if __has_builtin(__builtin_amdgcn_global_load_async_to_lds_b128)
#define HAVE_ASYNC_LDS 1
#endif
#endif
#ifndef HAVE_ASYNC_LDS
#define HAVE_ASYNC_LDS 0
#endif

#if HAVE_ASYNC_LDS
// builtin signature (from clang diagnostics):
//   (int4 addrspace(1)*, int4 addrspace(3)*, imm int offset, imm int cpol)
typedef int v4i_t __attribute__((ext_vector_type(4)));
typedef v4i_t __attribute__((address_space(1)))* async_gp_t;
typedef v4i_t __attribute__((address_space(3)))* async_lp_t;
#define ASYNC_B128(g, l) \
    __builtin_amdgcn_global_load_async_to_lds_b128((async_gp_t)(g), (async_lp_t)(l), 0, 0)
#if __has_builtin(__builtin_amdgcn_s_wait_asynccnt)
#define WAIT_ASYNC(n) __builtin_amdgcn_s_wait_asynccnt(n)
#else
#define WAIT_ASYNC(n) asm volatile("s_wait_asynccnt %0" :: "i"(n) : "memory")
#endif
#endif

__device__ __forceinline__ u16t f2bf(float f) {
    u32t u = __float_as_uint(f);
    u32t r = u + 0x7FFFu + ((u >> 16) & 1u);   // round-to-nearest-even
    return (u16t)(r >> 16);
}
__device__ __forceinline__ float bf2f(u16t h) {
    return __uint_as_float(((u32t)h) << 16);
}

#define GEMM_BM 128
#define GEMM_BN 128
#define GEMM_BK 32

// ---------------------------------------------------------------------------
// bf16 WMMA GEMM:  C[M,N] = A_bf16[M,K] @ Bt_bf16[N,K]^T  (B pre-transposed,
// so both operands are K-contiguous -> staging is pure 16-byte copies).
// Block tile 128x128, 8 waves in 4(M) x 2(N) grid, 32x64 per wave
// -> 8 WMMA per wave per K-step (64 flop per staged LDS byte).
// mode 0: store f32          mode 1: store bf16
// mode 2: store f32 + Res    mode 3: gelu -> bf16     mode 4: f32 + bias[n]
// ---------------------------------------------------------------------------
__global__ __launch_bounds__(256)
void gemm_bf16_wmma(const u16t* __restrict__ A, const u16t* __restrict__ Bt,
                    float* __restrict__ Cf, u16t* __restrict__ Cbf,
                    const float* __restrict__ Res, const float* __restrict__ bias,
                    int M, int N, int K, int mode)
{
    // double-buffered tiles, K-contiguous rows (row stride 32 elems = 64 B)
    __shared__ u16t As[2][GEMM_BM * GEMM_BK];   // [128][32]
    __shared__ u16t Bs[2][GEMM_BN * GEMM_BK];   // [128][32]

    const int t    = threadIdx.x;
    const int lane = t & 31;
    const int wave = t >> 5;
    const int hl   = lane >> 4;      // lane half: 0 or 1
    const int lrow = lane & 15;
    const int wm   = wave & 3;       // 0..3 : M sub-block (32 rows each)
    const int wn   = wave >> 2;      // 0..1 : N sub-block (64 cols each)
    const int blockN = blockIdx.x * GEMM_BN;
    const int blockM = blockIdx.y * GEMM_BM;

    v8f acc[2][4];
#pragma unroll
    for (int i = 0; i < 2; ++i)
#pragma unroll
        for (int j = 0; j < 4; ++j)
#pragma unroll
            for (int r = 0; r < 8; ++r) acc[i][j][r] = 0.0f;

    // ---- per-thread staging addresses (advance by 64 B per K-step) --------
    // A tile: 128 rows x 64 B -> 256 threads x 2 x b128
    const int srow = t >> 1, sh = t & 1;            // row 0..127, 32B half
    const char* aG = (const char*)A + ((size_t)(blockM + srow) * K + sh * 16) * 2;
    u16t* aL0 = &As[0][srow * 32 + sh * 16];
    u16t* aL1 = &As[1][srow * 32 + sh * 16];
    // B tile: 128 rows x 64 B -> 256 threads x 2 x b128
    const char* bG = (const char*)Bt + ((size_t)(blockN + srow) * K + sh * 16) * 2;
    u16t* bL0 = &Bs[0][srow * 32 + sh * 16];
    u16t* bL1 = &Bs[1][srow * 32 + sh * 16];

#if HAVE_ASYNC_LDS
    // ---- async (direct-to-LDS) double-buffered pipeline -------------------
    ASYNC_B128(aG, aL0);
    ASYNC_B128(aG + 32, aL0 + 16);
    ASYNC_B128(bG, bL0);
    ASYNC_B128(bG + 32, bL0 + 16);
    int buf = 0;
    for (int k0 = 0; k0 < K; k0 += GEMM_BK) {
        const bool more = (k0 + GEMM_BK) < K;
        if (more) {                 // prefetch next tile into other buffer
            const size_t kb = (size_t)(k0 + GEMM_BK) * 2;
            u16t* aL = buf ? aL0 : aL1;
            u16t* bL = buf ? bL0 : bL1;
            ASYNC_B128(aG + kb, aL);
            ASYNC_B128(aG + kb + 32, aL + 16);
            ASYNC_B128(bG + kb, bL);
            ASYNC_B128(bG + kb + 32, bL + 16);
            WAIT_ASYNC(4);          // current tile's 4 asyncs retired (in-order)
        } else {
            WAIT_ASYNC(0);
        }
        __syncthreads();

        union FU { v16bf v; u32t u[8]; };
        FU afr[2], bfr[4];
        const u32t* Asu = (const u32t*)As[buf];
        const u32t* Bsu = (const u32t*)Bs[buf];
#pragma unroll
        for (int i = 0; i < 2; ++i) {
            int m = wm * 32 + i * 16 + lrow;
#pragma unroll
            for (int e = 0; e < 8; ++e) {
                int kq = e + ((e >> 2) << 2) + hl * 4;
                afr[i].u[e] = Asu[m * 16 + kq];
            }
        }
#pragma unroll
        for (int j = 0; j < 4; ++j) {
            int n = wn * 64 + j * 16 + lrow;
#pragma unroll
            for (int e = 0; e < 8; ++e) {
                int kq = e + ((e >> 2) << 2) + hl * 4;
                bfr[j].u[e] = Bsu[n * 16 + kq];
            }
        }
#pragma unroll
        for (int i = 0; i < 2; ++i)
#pragma unroll
            for (int j = 0; j < 4; ++j)
                acc[i][j] = __builtin_amdgcn_wmma_f32_16x16x32_bf16(
                    false, afr[i].v, false, bfr[j].v,
                    (short)0, acc[i][j], false, false);
        __syncthreads();
        buf ^= 1;
    }
#else
    // ---- fallback: vector register staging (single buffer) ----------------
    for (int k0 = 0; k0 < K; k0 += GEMM_BK) {
        const size_t kb = (size_t)k0 * 2;
        uint4 av0 = *(const uint4*)(aG + kb);
        uint4 av1 = *(const uint4*)(aG + kb + 32);
        uint4 bv0 = *(const uint4*)(bG + kb);
        uint4 bv1 = *(const uint4*)(bG + kb + 32);
        *(uint4*)aL0 = av0;
        *(uint4*)(aL0 + 16) = av1;
        *(uint4*)bL0 = bv0;
        *(uint4*)(bL0 + 16) = bv1;
        __syncthreads();

        union FU { v16bf v; u32t u[8]; };
        FU afr[2], bfr[4];
        const u32t* Asu = (const u32t*)As[0];
        const u32t* Bsu = (const u32t*)Bs[0];
#pragma unroll
        for (int i = 0; i < 2; ++i) {
            int m = wm * 32 + i * 16 + lrow;
#pragma unroll
            for (int e = 0; e < 8; ++e) {
                int kq = e + ((e >> 2) << 2) + hl * 4;
                afr[i].u[e] = Asu[m * 16 + kq];
            }
        }
#pragma unroll
        for (int j = 0; j < 4; ++j) {
            int n = wn * 64 + j * 16 + lrow;
#pragma unroll
            for (int e = 0; e < 8; ++e) {
                int kq = e + ((e >> 2) << 2) + hl * 4;
                bfr[j].u[e] = Bsu[n * 16 + kq];
            }
        }
#pragma unroll
        for (int i = 0; i < 2; ++i)
#pragma unroll
            for (int j = 0; j < 4; ++j)
                acc[i][j] = __builtin_amdgcn_wmma_f32_16x16x32_bf16(
                    false, afr[i].v, false, bfr[j].v,
                    (short)0, acc[i][j], false, false);
        __syncthreads();
    }
    (void)aL1; (void)bL1;
#endif

    // --- epilogue ---
#pragma unroll
    for (int i = 0; i < 2; ++i) {
#pragma unroll
        for (int j = 0; j < 4; ++j) {
            int n = blockN + wn * 64 + j * 16 + lrow;
            int mbase = blockM + wm * 32 + i * 16 + hl * 8;
#pragma unroll
            for (int r = 0; r < 8; ++r) {
                size_t off = (size_t)(mbase + r) * N + n;
                float v = acc[i][j][r];
                if (mode == 2)      v += Res[off];
                else if (mode == 4) v += bias[n];
                if (mode == 3) {
                    v = 0.5f * v * (1.0f + erff(v * 0.70710678118654752f));
                    Cbf[off] = f2bf(v);
                } else if (mode == 1) {
                    Cbf[off] = f2bf(v);
                } else {
                    Cf[off] = v;
                }
            }
        }
    }
}

// ---------------------------------------------------------------------------
// Tiled fp32 -> bf16 transpose: W[K][N] -> Wt[N][K]. Coalesced both ways.
// ---------------------------------------------------------------------------
__global__ __launch_bounds__(256)
void cvt_transpose_bf16(const float* __restrict__ W, u16t* __restrict__ Wt,
                        int K, int N)
{
    __shared__ float tile[32][33];
    const int k0 = blockIdx.y * 32, n0 = blockIdx.x * 32;
    const int tx = threadIdx.x & 31, ty = threadIdx.x >> 5;   // ty: 0..7
#pragma unroll
    for (int r = 0; r < 32; r += 8)
        tile[ty + r][tx] = W[(size_t)(k0 + ty + r) * N + n0 + tx];
    __syncthreads();
#pragma unroll
    for (int r = 0; r < 32; r += 8)
        Wt[(size_t)(n0 + ty + r) * K + k0 + tx] = f2bf(tile[tx][ty + r]);
}

// ---------------------------------------------------------------------------
// LayerNorm over D=1024. One block per row (256 threads, 4 elems/thread).
// dst row = dbase + (r/rpb)*bstride + (r%rpb)
// ---------------------------------------------------------------------------
__global__ __launch_bounds__(256)
void ln_kernel(const float* __restrict__ src, const float* __restrict__ g,
               const float* __restrict__ bb, u16t* __restrict__ dst_bf,
               float* __restrict__ dst_f, int rpb, int bstride, int dbase)
{
    __shared__ float red[256];
    const int r = blockIdx.x, t = threadIdx.x;
    const float4 xv = ((const float4*)(src + (size_t)r * 1024))[t];

    red[t] = xv.x + xv.y + xv.z + xv.w;
    __syncthreads();
    for (int st = 128; st > 0; st >>= 1) {
        if (t < st) red[t] += red[t + st];
        __syncthreads();
    }
    const float mean = red[0] * (1.0f / 1024.0f);
    __syncthreads();

    float dx = xv.x - mean, dy = xv.y - mean, dz = xv.z - mean, dw = xv.w - mean;
    red[t] = dx * dx + dy * dy + dz * dz + dw * dw;
    __syncthreads();
    for (int st = 128; st > 0; st >>= 1) {
        if (t < st) red[t] += red[t + st];
        __syncthreads();
    }
    const float rstd = rsqrtf(red[0] * (1.0f / 1024.0f) + 1e-5f);

    const float4 gv = ((const float4*)g)[t];
    const float4 bv = ((const float4*)bb)[t];
    float y0 = dx * rstd * gv.x + bv.x;
    float y1 = dy * rstd * gv.y + bv.y;
    float y2 = dz * rstd * gv.z + bv.z;
    float y3 = dw * rstd * gv.w + bv.w;

    const size_t drow = (size_t)dbase + (size_t)(r / rpb) * bstride + (r % rpb);
    const size_t c = drow * 1024 + (size_t)t * 4;
    if (dst_bf) {
        dst_bf[c + 0] = f2bf(y0); dst_bf[c + 1] = f2bf(y1);
        dst_bf[c + 2] = f2bf(y2); dst_bf[c + 3] = f2bf(y3);
    }
    if (dst_f) {
        float4 yv = make_float4(y0, y1, y2, y3);
        ((float4*)dst_f)[c / 4] = yv;
    }
}

// ---------------------------------------------------------------------------
// Masked cross-attention, online softmax. One block per (b,h).
// q: fp32 (B,64,H*64)  kv: bf16 (B,2112,2048) [k | v]   o: bf16 (B,64,H*64)
// ---------------------------------------------------------------------------
__global__ __launch_bounds__(256)
void attn_kernel(const float* __restrict__ q, const u16t* __restrict__ kv,
                 const int* __restrict__ mask, u16t* __restrict__ o)
{
    __shared__ u16t ksm[64 * 64];
    __shared__ u16t vsm[64 * 64];
    const int b = blockIdx.y, h = blockIdx.x;
    const int t = threadIdx.x;
    const int qi = t >> 2, sub = t & 3;
    const float scale = 0.125f;   // 64^-0.5

    float qr[16];
#pragma unroll
    for (int d = 0; d < 16; ++d)
        qr[d] = q[(size_t)(b * 64 + qi) * 1024 + h * 64 + sub * 16 + d] * scale;

    float oacc[16];
#pragma unroll
    for (int d = 0; d < 16; ++d) oacc[d] = 0.0f;
    float mrun = -1e30f, ssum = 0.0f;

    const u32t* kvu = (const u32t*)kv;
    u32t* ku = (u32t*)ksm;
    u32t* vu = (u32t*)vsm;

    for (int c = 0; c < 33; ++c) {          // 2112 = 33 * 64 keys
        const int j0 = c * 64;
#pragma unroll
        for (int w = 0; w < 8; ++w) {
            int idx = t + 256 * w;
            int row = idx >> 5, cp = idx & 31;
            size_t base = (size_t)(b * 2112 + j0 + row) * 1024 + h * 32;
            ku[idx] = kvu[base + cp];
            vu[idx] = kvu[base + 512 + cp];
        }
        __syncthreads();

        for (int j = 0; j < 64; ++j) {
            float s = 0.0f;
            const u16t* kr = &ksm[j * 64 + sub * 16];
#pragma unroll
            for (int d = 0; d < 16; ++d) s += qr[d] * bf2f(kr[d]);
            s += __shfl_xor(s, 1, 32);
            s += __shfl_xor(s, 2, 32);

            const int jj = j0 + j;
            const bool keep = (jj >= 2048) || (mask[b * 2048 + jj] > 0);
            if (!keep) s = -30000.0f;       // exp underflows to 0 vs real max

            const float mnew = fmaxf(mrun, s);
            const float corr = __expf(mrun - mnew);
            const float pw   = __expf(s - mnew);
            ssum = ssum * corr + pw;
            const u16t* vr = &vsm[j * 64 + sub * 16];
#pragma unroll
            for (int d = 0; d < 16; ++d)
                oacc[d] = oacc[d] * corr + pw * bf2f(vr[d]);
            mrun = mnew;
        }
        __syncthreads();
    }

    const float inv = 1.0f / ssum;
#pragma unroll
    for (int d = 0; d < 16; ++d)
        o[(size_t)(b * 64 + qi) * 1024 + h * 64 + sub * 16 + d] = f2bf(oacc[d] * inv);
}

// ---------------------------------------------------------------------------
// Small helpers
// ---------------------------------------------------------------------------
__global__ __launch_bounds__(256)
void cvt_f32_bf16(const float* __restrict__ src, u16t* __restrict__ dst, int n)
{
    int i = blockIdx.x * 256 + threadIdx.x;
    if (i < n) dst[i] = f2bf(src[i]);
}

__global__ __launch_bounds__(256)
void bcast_lat(const float* __restrict__ lat0, float* __restrict__ dst,
               int n, int per)
{
    int i = blockIdx.x * 256 + threadIdx.x;
    if (i < n) dst[i] = lat0[i % per];
}

// ---------------------------------------------------------------------------
// Orchestration
// ---------------------------------------------------------------------------
extern "C" void kernel_launch(void* const* d_in, const int* in_sizes, int n_in,
                              void* d_out, int out_size, void* d_ws, size_t ws_size,
                              hipStream_t stream)
{
    (void)in_sizes; (void)n_in; (void)out_size; (void)ws_size;

    const int B = 16, N = 2048, D = 1024, DEPTH = 6;
    const int NL = 64, INNER = 1024, FF = 4096, NK = N + NL; // 2112

    const float* embeds     = (const float*)d_in[0];
    const float* latents    = (const float*)d_in[1];
    const float* Wq         = (const float*)d_in[2];
    const float* Wkv        = (const float*)d_in[3];
    const float* Wo         = (const float*)d_in[4];
    const float* W1         = (const float*)d_in[5];
    const float* W2         = (const float*)d_in[6];
    const float* ln_media_g = (const float*)d_in[7];
    const float* ln_media_b = (const float*)d_in[8];
    const float* ln_lat_g   = (const float*)d_in[9];
    const float* ln_lat_b   = (const float*)d_in[10];
    const float* ff_ln_g    = (const float*)d_in[11];
    const float* ff_ln_b    = (const float*)d_in[12];
    const float* proj_W     = (const float*)d_in[13];
    const float* proj_b     = (const float*)d_in[14];
    const float* final_g    = (const float*)d_in[15];
    const float* final_b    = (const float*)d_in[16];
    const int*   maskp      = (const int*)d_in[17];
    float* out = (float*)d_out;

    // workspace carve-up
    char* ws = (char*)d_ws;
    size_t off = 0;
    auto carve = [&](size_t bytes) -> char* {
        char* p = ws + off;
        off += (bytes + 255) & ~(size_t)255;
        return p;
    };
    u16t* xcat_bf = (u16t*)carve((size_t)B * NK * D * 2);     // ln(media)||ln(lat)
    u16t* kv_bf   = (u16t*)carve((size_t)B * NK * 2 * INNER * 2);
    float* lat_f  = (float*)carve((size_t)B * NL * D * 4);
    u16t* l_bf    = (u16t*)carve((size_t)B * NL * D * 2);
    float* q_f    = (float*)carve((size_t)B * NL * INNER * 4);
    u16t* o_bf    = (u16t*)carve((size_t)B * NL * INNER * 2);
    u16t* hln_bf  = (u16t*)carve((size_t)B * NL * D * 2);
    u16t* t1_bf   = (u16t*)carve((size_t)B * NL * FF * 2);
    u16t* lat_bf  = (u16t*)carve((size_t)B * NL * D * 2);
    float* proj_f = (float*)carve((size_t)B * NL * D * 4);
    u16t* wstage  = (u16t*)carve((size_t)D * FF * 2);         // transposed weight

    const int ML = B * NL;  // 1024 latent rows

    // lat = broadcast(latents)
    bcast_lat<<<(ML * D) / 256, 256, 0, stream>>>(latents, lat_f, ML * D, NL * D);

    for (int i = 0; i < DEPTH; ++i) {
        const float* lmg = ln_media_g + (size_t)i * D;
        const float* lmb = ln_media_b + (size_t)i * D;
        const float* llg = ln_lat_g + (size_t)i * D;
        const float* llb = ln_lat_b + (size_t)i * D;

        // x = ln(embeds) -> xcat rows [0,2048) per batch
        ln_kernel<<<B * N, 256, 0, stream>>>(embeds, lmg, lmb, xcat_bf, nullptr,
                                             N, NK, 0);
        // l = ln(lat) -> xcat rows [2048,2112) per batch, and contiguous l_bf
        ln_kernel<<<ML, 256, 0, stream>>>(lat_f, llg, llb, xcat_bf, nullptr,
                                          NL, NK, N);
        ln_kernel<<<ML, 256, 0, stream>>>(lat_f, llg, llb, l_bf, nullptr,
                                          NL, NL, 0);

        // q = l @ Wq[i]
        cvt_transpose_bf16<<<dim3(INNER / 32, D / 32), 256, 0, stream>>>(
            Wq + (size_t)i * D * INNER, wstage, D, INNER);
        gemm_bf16_wmma<<<dim3(INNER / GEMM_BN, ML / GEMM_BM), 256, 0, stream>>>(
            l_bf, wstage, q_f, nullptr, nullptr, nullptr, ML, INNER, D, 0);

        // kv = xcat @ Wkv[i]   -- dominant GEMM
        cvt_transpose_bf16<<<dim3((2 * INNER) / 32, D / 32), 256, 0, stream>>>(
            Wkv + (size_t)i * D * 2 * INNER, wstage, D, 2 * INNER);
        gemm_bf16_wmma<<<dim3((2 * INNER) / GEMM_BN, (B * NK) / GEMM_BM), 256, 0, stream>>>(
            xcat_bf, wstage, nullptr, kv_bf, nullptr, nullptr,
            B * NK, 2 * INNER, D, 1);

        // o = softmax(q k^T + mask) v
        attn_kernel<<<dim3(16, B), 256, 0, stream>>>(q_f, kv_bf, maskp, o_bf);

        // lat = o @ Wo[i] + lat
        cvt_transpose_bf16<<<dim3(D / 32, INNER / 32), 256, 0, stream>>>(
            Wo + (size_t)i * INNER * D, wstage, INNER, D);
        gemm_bf16_wmma<<<dim3(D / GEMM_BN, ML / GEMM_BM), 256, 0, stream>>>(
            o_bf, wstage, lat_f, nullptr, lat_f, nullptr, ML, D, INNER, 2);

        // h = ln(lat); t1 = gelu(h @ W1[i]); lat = t1 @ W2[i] + lat
        ln_kernel<<<ML, 256, 0, stream>>>(lat_f, ff_ln_g + (size_t)i * D,
                                          ff_ln_b + (size_t)i * D,
                                          hln_bf, nullptr, NL, NL, 0);
        cvt_transpose_bf16<<<dim3(FF / 32, D / 32), 256, 0, stream>>>(
            W1 + (size_t)i * D * FF, wstage, D, FF);
        gemm_bf16_wmma<<<dim3(FF / GEMM_BN, ML / GEMM_BM), 256, 0, stream>>>(
            hln_bf, wstage, nullptr, t1_bf, nullptr, nullptr, ML, FF, D, 3);
        cvt_transpose_bf16<<<dim3(D / 32, FF / 32), 256, 0, stream>>>(
            W2 + (size_t)i * FF * D, wstage, FF, D);
        gemm_bf16_wmma<<<dim3(D / GEMM_BN, ML / GEMM_BM), 256, 0, stream>>>(
            t1_bf, wstage, lat_f, nullptr, lat_f, nullptr, ML, D, FF, 2);
    }

    // out = ln(lat @ proj_W + proj_b)
    cvt_f32_bf16<<<(ML * D) / 256, 256, 0, stream>>>(lat_f, lat_bf, ML * D);
    cvt_transpose_bf16<<<dim3(D / 32, D / 32), 256, 0, stream>>>(
        proj_W, wstage, D, D);
    gemm_bf16_wmma<<<dim3(D / GEMM_BN, ML / GEMM_BM), 256, 0, stream>>>(
        lat_bf, wstage, proj_f, nullptr, nullptr, proj_b, ML, D, D, 4);
    ln_kernel<<<ML, 256, 0, stream>>>(proj_f, final_g, final_b, nullptr, out,
                                      NL, NL, 0);
}
